// MultiHeadAttention_11046655885758
// MI455X (gfx1250) — compile-verified, matrix operand order checked by reading, not run on hardware
//
#include <hip/hip_runtime.h>

// ---------------------------------------------------------------------------
// Problem constants (from reference)
// ---------------------------------------------------------------------------
constexpr int cB  = 2;
constexpr int cT  = 4096;
constexpr int cD  = 1024;
constexpr int cH  = 16;
constexpr int cDK = 64;
constexpr int cBS = 256;
constexpr int cNB = 16;     // T / BS
constexpr int cW3 = 768;    // 3*BS window

// ---------------------------------------------------------------------------
// CDNA5 WMMA plumbing (wave32, 16x16x32 bf16 -> f32)
// ---------------------------------------------------------------------------
typedef __attribute__((ext_vector_type(16))) __bf16 v16bf;
typedef __attribute__((ext_vector_type(8)))  float  v8f;

struct Pair128 { uint4 a, b; };   // 32 bytes == one 16x32 bf16 fragment

__device__ __forceinline__ unsigned short f2bf(float f) {
  // round-to-nearest-even fp32 -> bf16 (used only OUTSIDE hot loops)
  unsigned int u = __float_as_uint(f);
  unsigned int r = u + 0x7FFFu + ((u >> 16) & 1u);
  return (unsigned short)(r >> 16);
}
__device__ __forceinline__ float bf2f(unsigned short h) {
  return __uint_as_float(((unsigned int)h) << 16);
}

// Fragment load: CDNA5 16-bit A/B layout puts frag elements 0..7 at
// K = half*8 .. half*8+7 and elements 8..15 at K = 16+half*8 .. +7,
// i.e. two contiguous 16-byte runs -> two b128 loads (global or LDS).
__device__ __forceinline__ v16bf fragld(const unsigned short* p, int half) {
  Pair128 pr;
  pr.a = *(const uint4*)(p + half * 8);
  pr.b = *(const uint4*)(p + 16 + half * 8);
  return __builtin_bit_cast(v16bf, pr);
}
__device__ __forceinline__ v16bf frag_zero() {
  Pair128 pr{};
  return __builtin_bit_cast(v16bf, pr);
}

__device__ __forceinline__ v8f wmma_bf(v16bf a, v16bf b, v8f c) {
  return __builtin_amdgcn_wmma_f32_16x16x32_bf16(false, a, false, b,
                                                 (short)0, c, false, false);
}
// split-bf16 error-compensated product: (ah+al)*(bh+bl) ~ ah*bh + ah*bl + al*bh
__device__ __forceinline__ v8f wmma3(v16bf ah, v16bf al, v16bf bh, v16bf bl, v8f c) {
  c = wmma_bf(ah, bh, c);
  c = wmma_bf(ah, bl, c);
  c = wmma_bf(al, bh, c);
  return c;
}

// ---------------------------------------------------------------------------
// One-time fp32 -> bf16 hi/lo split kernels
// ---------------------------------------------------------------------------
__global__ __launch_bounds__(256) void split_rm(const float* __restrict__ X,
                                                unsigned short* __restrict__ H,
                                                unsigned short* __restrict__ L) {
  const int i = blockIdx.x * 256 + threadIdx.x;
  const float x = X[i];
  const unsigned short h = f2bf(x);
  H[i] = h;
  L[i] = f2bf(x - bf2f(h));
}

// W [K][N] -> Wt hi/lo [N][K] (so B-fragment K-dim is contiguous)
__global__ __launch_bounds__(256) void split_tr(const float* __restrict__ W,
                                                unsigned short* __restrict__ H,
                                                unsigned short* __restrict__ L) {
  const int i = blockIdx.x * 256 + threadIdx.x;   // coalesced read over K*N
  const int kk = i / cD;
  const int nn = i % cD;
  const float x = W[i];
  const unsigned short h = f2bf(x);
  const size_t o = (size_t)nn * cD + kk;
  H[o] = h;
  L[o] = f2bf(x - bf2f(h));
}

// ---------------------------------------------------------------------------
// Projection GEMM on pre-split bf16 hi/lo inputs.
//   A: [M][1024] split row-major;  Wt: [1024 N][1024 K] split.
// Block = 256 threads (8 waves stacked along M); block computes 128(M)x64(N).
// The shared 64x32 W hi/lo tile is staged in LDS (8KB/step, double-buffered,
// one __syncthreads per K-step); B-fragments come from ds_load_b128.
// MODE 0: write split row-major (q, k)
// MODE 1: write split transposed per head [b][h][d][t] (v)
// MODE 2: write fp32 row-major (final output)
// ---------------------------------------------------------------------------
template <int MODE>
__global__ __launch_bounds__(256) void proj_gemm(
    const unsigned short* __restrict__ Ah, const unsigned short* __restrict__ Al,
    const unsigned short* __restrict__ Wh, const unsigned short* __restrict__ Wl,
    float* __restrict__ Cf,
    unsigned short* __restrict__ Ch, unsigned short* __restrict__ Cl) {
  constexpr int K = cD, N = cD;
  constexpr int NSTEP = K / 32;
  // [buf][hi/lo][col 64][k 32]
  __shared__ unsigned short wtile[2 * 2 * 64 * 32];   // 16KB

  const int tid  = threadIdx.x;
  const int wave = tid >> 5;
  const int lane = tid & 31;
  const int half = lane >> 4;
  const int ln   = lane & 15;

  const int m0 = (blockIdx.x * 8 + wave) * 16;
  const int n0 = blockIdx.y * 64;

  v8f acc[4] = {};

  const unsigned short* arh = Ah + (size_t)(m0 + ln) * K;
  const unsigned short* arl = Al + (size_t)(m0 + ln) * K;

  // cooperative stage of one 64x32 hi+lo W tile: 256 threads x 2 b128 chunks
  auto stage = [&](int step) {
    const int buf = step & 1;
    const int kb  = step * 32;
#pragma unroll
    for (int c = 0; c < 2; ++c) {
      const int chunk = tid * 2 + c;        // 0..511
      const int hl    = chunk >> 8;         // 0 = hi, 1 = lo
      const int col   = (chunk & 255) >> 2; // 0..63
      const int kk8   = (chunk & 3) * 8;    // 0,8,16,24
      const unsigned short* src =
          (hl ? Wl : Wh) + (size_t)(n0 + col) * K + kb + kk8;
      *(uint4*)&wtile[(((buf * 2 + hl) * 64) + col) * 32 + kk8] =
          *(const uint4*)src;
    }
  };

  stage(0);
  for (int step = 0; step < NSTEP; ++step) {
    __syncthreads();
    if (step + 1 < NSTEP) stage(step + 1);   // overlaps this step's WMMAs

    const int kb = step * 32;
    const v16bf a_hi = fragld(arh + kb, half);
    const v16bf a_lo = fragld(arl + kb, half);
    __builtin_prefetch(arh + kb + 32, 0, 1);   // global_prefetch_b8

    const unsigned short* wt = &wtile[(step & 1) * 2 * 64 * 32];
#pragma unroll
    for (int sub = 0; sub < 4; ++sub) {
      const unsigned short* colp = wt + (sub * 16 + ln) * 32;
      const v16bf b_hi = fragld(colp, half);
      const v16bf b_lo = fragld(colp + 64 * 32, half);
      acc[sub] = wmma3(a_hi, a_lo, b_hi, b_lo, acc[sub]);
    }
  }

#pragma unroll
  for (int sub = 0; sub < 4; ++sub) {
#pragma unroll
    for (int r = 0; r < 8; ++r) {
      const int   m   = m0 + r + 8 * half;
      const int   col = n0 + sub * 16 + ln;
      const float val = acc[sub][r];
      if constexpr (MODE == 2) {
        Cf[(size_t)m * N + col] = val;
      } else {
        const unsigned short h0 = f2bf(val);
        const unsigned short l0 = f2bf(val - bf2f(h0));
        if constexpr (MODE == 0) {
          const size_t idx = (size_t)m * N + col;
          Ch[idx] = h0;
          Cl[idx] = l0;
        } else {  // MODE 1: v transposed per head [b][h][d][t]
          const int bb = m >> 12;            // m / T
          const int t  = m & (cT - 1);
          const int hh = col >> 6;           // col / DK
          const int dd = col & (cDK - 1);
          const size_t idx = ((size_t)((bb * cH + hh) * cDK + dd)) * cT + t;
          Ch[idx] = h0;
          Cl[idx] = l0;
        }
      }
    }
  }
}

// ---------------------------------------------------------------------------
// Fused blocked attention with talking heads (two-pass flash style).
// Grid = (qtile=16, n=NB, b=B). Block = 512 threads = 16 waves; wave w is both
// score producer for head h=w and mixed consumer for head g=w.
// LDS scoreBuf layout: [q16][kc64][h16] f32 so head-mixing reads 16 contiguous
// floats (ds_load_b128 x4). Same 64KB later reused for P and for per-wave bf16
// hi/lo staging of P' (wave h owns region h*4KB).
// ---------------------------------------------------------------------------
__global__ __launch_bounds__(512) void attn_talking_heads(
    const unsigned short* __restrict__ qh, const unsigned short* __restrict__ ql,
    const unsigned short* __restrict__ kh, const unsigned short* __restrict__ kl,
    const unsigned short* __restrict__ vth, const unsigned short* __restrict__ vtl,
    const unsigned char* __restrict__ qmask,
    const float* __restrict__ preW, const float* __restrict__ preB,
    const float* __restrict__ postW, const float* __restrict__ postB,
    const float* __restrict__ relP, const float* __restrict__ absP,
    unsigned short* __restrict__ aoh, unsigned short* __restrict__ aol) {
  __shared__ float scoreBuf[16 * 64 * cH];   // 64KB

  const int tid  = threadIdx.x;
  const int wave = tid >> 5;   // == head h == out-head g
  const int lane = tid & 31;
  const int half = lane >> 4;
  const int ln   = lane & 15;
  const int h    = wave;

  const int qt = blockIdx.x;
  const int n  = blockIdx.y;
  const int b  = blockIdx.z;

  const float rel  = relP[0];
  const float asub = absP[0];

  float preRow[16], postRow[16];
#pragma unroll
  for (int j = 0; j < 16; ++j) {
    preRow[j]  = preW[h * 16 + j];
    postRow[j] = postW[h * 16 + j];
  }
  const float pbias = preB[h];
  const float obias = postB[h];

  const int qrow0 = n * cBS + qt * 16;   // global t of q row 0 of this tile
  const int wbase = n * cBS - cBS;       // global t of window col 0

  // hoisted q_mask for the 8 rows this lane covers
  bool qmv[8];
#pragma unroll
  for (int r = 0; r < 8; ++r)
    qmv[r] = (qmask[b * cT + qrow0 + r + 8 * half] != 0);

  // ---- Q fragments for head h: 16x64 -> two 16x32 A-frags, hi/lo split ----
  v16bf qa_hi[2], qa_lo[2];
  {
    const size_t base = ((size_t)(b * cT + qrow0 + ln)) * cD + h * cDK;
#pragma unroll
    for (int ks = 0; ks < 2; ++ks) {
      qa_hi[ks] = fragld(qh + base + ks * 32, half);
      qa_lo[ks] = fragld(ql + base + ks * 32, half);
    }
  }

  float m[8], l[8];
#pragma unroll
  for (int r = 0; r < 8; ++r) { m[r] = -3.0e38f; l[r] = 0.0f; }
  v8f oacc[4] = {};

  for (int pass = 0; pass < 2; ++pass) {
    for (int kt = 0; kt < cW3 / 64; ++kt) {
      const int ktbase = kt * 64;

      // ---- 1) raw scores for head h over (16q x 64kc), WMMA ----
#pragma unroll
      for (int sub = 0; sub < 4; ++sub) {
        v8f acc = {};
        const int c   = ktbase + sub * 16 + ln;   // window column (N of WMMA)
        const int pos = wbase + c;                // global t of that column
        const bool inT = (pos >= 0) && (pos < cT);
        const size_t kbase = inT ? ((size_t)(b * cT + pos)) * cD + h * cDK : 0;
#pragma unroll
        for (int ks = 0; ks < 2; ++ks) {
          const v16bf b_hi = inT ? fragld(kh + kbase + ks * 32, half) : frag_zero();
          const v16bf b_lo = inT ? fragld(kl + kbase + ks * 32, half) : frag_zero();
          acc = wmma3(qa_hi[ks], qa_lo[ks], b_hi, b_lo, acc);
        }
        // scale by 1/sqrt(DK), distance damping, abs_sub -> LDS
#pragma unroll
        for (int r = 0; r < 8; ++r) {
          const int qrow = r + 8 * half;
          const int rr   = qt * 16 + qrow;
          float s = acc[r] * 0.125f;
          const float dist = fabsf((float)(cBS + rr - c));
          s = s / (1.0f + dist * rel);
          if (!(c >= rr + cBS)) s -= asub;
          scoreBuf[(qrow * 64 + sub * 16 + ln) * 16 + h] = s;
        }
      }
      __syncthreads();

      // ---- 2) pre-talk mix across heads for g = wave, then mask ----
      float Sg[4][8];
#pragma unroll
      for (int sub = 0; sub < 4; ++sub) {
        const int c   = ktbase + sub * 16 + ln;
        const int pos = wbase + c;
#pragma unroll
        for (int r = 0; r < 8; ++r) {
          const int qrow = r + 8 * half;
          const float4* sp = (const float4*)&scoreBuf[(qrow * 64 + sub * 16 + ln) * 16];
          const float4 s0 = sp[0], s1 = sp[1], s2 = sp[2], s3 = sp[3];
          float acc = pbias;
          acc += preRow[0] * s0.x + preRow[1] * s0.y + preRow[2] * s0.z + preRow[3] * s0.w;
          acc += preRow[4] * s1.x + preRow[5] * s1.y + preRow[6] * s1.z + preRow[7] * s1.w;
          acc += preRow[8] * s2.x + preRow[9] * s2.y + preRow[10] * s2.z + preRow[11] * s2.w;
          acc += preRow[12] * s3.x + preRow[13] * s3.y + preRow[14] * s3.z + preRow[15] * s3.w;
          const int rr = qt * 16 + qrow;
          const bool ok = (c <= rr + cBS) && (pos >= 0) && (pos < cT) && qmv[r];
          Sg[sub][r] = ok ? acc : -1.0e13f;
        }
      }

      if (pass == 0) {
        // ---- online row max / sum-exp (width-16 butterfly) ----
#pragma unroll
        for (int r = 0; r < 8; ++r) {
          float tmax = Sg[0][r];
#pragma unroll
          for (int sub = 1; sub < 4; ++sub) tmax = fmaxf(tmax, Sg[sub][r]);
#pragma unroll
          for (int off = 1; off < 16; off <<= 1)
            tmax = fmaxf(tmax, __shfl_xor(tmax, off, 16));
          const float nm = fmaxf(m[r], tmax);
          float ss = 0.0f;
#pragma unroll
          for (int sub = 0; sub < 4; ++sub) ss += __expf(Sg[sub][r] - nm);
#pragma unroll
          for (int off = 1; off < 16; off <<= 1) ss += __shfl_xor(ss, off, 16);
          l[r] = l[r] * __expf(m[r] - nm) + ss;
          m[r] = nm;
        }
        __syncthreads();   // all mix-reads done before next tile's writes
      } else {
        __syncthreads();   // all mix-reads of raw scores complete
        // ---- 3) normalized P_h back into LDS ----
#pragma unroll
        for (int sub = 0; sub < 4; ++sub)
#pragma unroll
          for (int r = 0; r < 8; ++r) {
            const int qrow = r + 8 * half;
            const float p = __expf(Sg[sub][r] - m[r]) / l[r];
            scoreBuf[(qrow * 64 + sub * 16 + ln) * 16 + h] = p;
          }
        __syncthreads();
        // ---- 4) post-talk mix: P'_g = sum_h W[g,h] P_h + b_g (bias on ALL k;
        //         OOB window columns read V as zero, matching the zero-pad) ---
        float Pp[4][8];
#pragma unroll
        for (int sub = 0; sub < 4; ++sub)
#pragma unroll
          for (int r = 0; r < 8; ++r) {
            const int qrow = r + 8 * half;
            const float4* sp = (const float4*)&scoreBuf[(qrow * 64 + sub * 16 + ln) * 16];
            const float4 s0 = sp[0], s1 = sp[1], s2 = sp[2], s3 = sp[3];
            float acc = obias;
            acc += postRow[0] * s0.x + postRow[1] * s0.y + postRow[2] * s0.z + postRow[3] * s0.w;
            acc += postRow[4] * s1.x + postRow[5] * s1.y + postRow[6] * s1.z + postRow[7] * s1.w;
            acc += postRow[8] * s2.x + postRow[9] * s2.y + postRow[10] * s2.z + postRow[11] * s2.w;
            acc += postRow[12] * s3.x + postRow[13] * s3.y + postRow[14] * s3.z + postRow[15] * s3.w;
            Pp[sub][r] = acc;
          }
        __syncthreads();   // P reads done; safe to reuse LDS for staging
        // ---- 5) bf16 hi/lo staging of P' in this wave's own 4KB region ----
        unsigned short* stg = (unsigned short*)scoreBuf + h * 2048;
#pragma unroll
        for (int sub = 0; sub < 4; ++sub)
#pragma unroll
          for (int r = 0; r < 8; ++r) {
            const int qrow = r + 8 * half;
            const unsigned short hi = f2bf(Pp[sub][r]);
            const unsigned short lo = f2bf(Pp[sub][r] - bf2f(hi));
            stg[qrow * 64 + sub * 16 + ln]        = hi;
            stg[1024 + qrow * 64 + sub * 16 + ln] = lo;
          }
        // ---- 6) P' A-fragments from staging (b128 LDS reads, same wave) ----
        v16bf pa_hi[2], pa_lo[2];
#pragma unroll
        for (int ks = 0; ks < 2; ++ks) {
          pa_hi[ks] = fragld(stg + ln * 64 + ks * 32, half);
          pa_lo[ks] = fragld(stg + 1024 + ln * 64 + ks * 32, half);
        }
        // ---- 7) O += P'(16x64) x V(64x64); V from transposed split [d][t] ---
#pragma unroll
        for (int dsub = 0; dsub < 4; ++dsub) {
          v8f acc = oacc[dsub];
          const int dcol = dsub * 16 + ln;
          const size_t vrow = ((size_t)((b * cH + h) * cDK + dcol)) * cT;
#pragma unroll
          for (int ks = 0; ks < 2; ++ks) {
            const int tA = wbase + ktbase + ks * 32 + half * 8;  // 8-aligned span
            const int tB = tA + 16;
            Pair128 ph{}, pl{};
            if (tA >= 0 && tA < cT) {
              ph.a = *(const uint4*)(vth + vrow + tA);
              pl.a = *(const uint4*)(vtl + vrow + tA);
            }
            if (tB >= 0 && tB < cT) {
              ph.b = *(const uint4*)(vth + vrow + tB);
              pl.b = *(const uint4*)(vtl + vrow + tB);
            }
            acc = wmma3(pa_hi[ks], pa_lo[ks],
                        __builtin_bit_cast(v16bf, ph),
                        __builtin_bit_cast(v16bf, pl), acc);
          }
          oacc[dsub] = acc;
        }
        __syncthreads();   // staging/raw region reused next tile
      }
    } // k-tiles
  }   // passes

  // ---- write O split (B,T,H*DK) for the final Wo GEMM ----
#pragma unroll
  for (int dsub = 0; dsub < 4; ++dsub)
#pragma unroll
    for (int r = 0; r < 8; ++r) {
      const int t = qrow0 + r + 8 * half;
      const size_t idx = ((size_t)(b * cT + t)) * cD + h * cDK + dsub * 16 + ln;
      const float val = oacc[dsub][r];
      const unsigned short h0 = f2bf(val);
      aoh[idx] = h0;
      aol[idx] = f2bf(val - bf2f(h0));
    }
}

// ---------------------------------------------------------------------------
// Launch
// ---------------------------------------------------------------------------
extern "C" void kernel_launch(void* const* d_in, const int* in_sizes, int n_in,
                              void* d_out, int out_size, void* d_ws,
                              size_t ws_size, hipStream_t stream) {
  const float*         x     = (const float*)d_in[0];
  const unsigned char* qmask = (const unsigned char*)d_in[1];
  const float*         Wq    = (const float*)d_in[2];
  const float*         Wk    = (const float*)d_in[3];
  const float*         Wv    = (const float*)d_in[4];
  const float*         Wo    = (const float*)d_in[5];
  const float*         rel   = (const float*)d_in[6];
  const float*         asub  = (const float*)d_in[7];
  const float*         preW  = (const float*)d_in[8];
  const float*         preB  = (const float*)d_in[9];
  const float*         postW = (const float*)d_in[10];
  const float*         postB = (const float*)d_in[11];

  const size_t SZ  = (size_t)cB * cT * cD;  // 8,388,608 activation elements
  const size_t WSZ = (size_t)cD * cD;       // 1,048,576 weight elements

  unsigned short* p = (unsigned short*)d_ws;
  unsigned short* xs_h = p;            p += SZ;
  unsigned short* xs_l = p;            p += SZ;
  unsigned short* wq_h = p;            p += WSZ;
  unsigned short* wq_l = p;            p += WSZ;
  unsigned short* wk_h = p;            p += WSZ;
  unsigned short* wk_l = p;            p += WSZ;
  unsigned short* wv_h = p;            p += WSZ;
  unsigned short* wv_l = p;            p += WSZ;
  unsigned short* wo_h = p;            p += WSZ;
  unsigned short* wo_l = p;            p += WSZ;
  unsigned short* qs_h = p;            p += SZ;
  unsigned short* qs_l = p;            p += SZ;
  unsigned short* ks_h = p;            p += SZ;
  unsigned short* ks_l = p;            p += SZ;
  unsigned short* vt_h = p;            p += SZ;
  unsigned short* vt_l = p;            p += SZ;
  unsigned short* ao_h = p;            p += SZ;
  unsigned short* ao_l = p;            p += SZ;

  // one-time precision split (+ weight transpose to [N][K])
  split_rm<<<dim3((unsigned)(SZ / 256)), 256, 0, stream>>>(x, xs_h, xs_l);
  split_tr<<<dim3((unsigned)(WSZ / 256)), 256, 0, stream>>>(Wq, wq_h, wq_l);
  split_tr<<<dim3((unsigned)(WSZ / 256)), 256, 0, stream>>>(Wk, wk_h, wk_l);
  split_tr<<<dim3((unsigned)(WSZ / 256)), 256, 0, stream>>>(Wv, wv_h, wv_l);
  split_tr<<<dim3((unsigned)(WSZ / 256)), 256, 0, stream>>>(Wo, wo_h, wo_l);

  const dim3 gblk(256);
  const dim3 ggrid(cB * cT / 128, cD / 64, 1);

  proj_gemm<0><<<ggrid, gblk, 0, stream>>>(xs_h, xs_l, wq_h, wq_l,
                                           nullptr, qs_h, qs_l);
  proj_gemm<0><<<ggrid, gblk, 0, stream>>>(xs_h, xs_l, wk_h, wk_l,
                                           nullptr, ks_h, ks_l);
  proj_gemm<1><<<ggrid, gblk, 0, stream>>>(xs_h, xs_l, wv_h, wv_l,
                                           nullptr, vt_h, vt_l);

  attn_talking_heads<<<dim3(cBS / 16, cNB, cB), 512, 0, stream>>>(
      qs_h, qs_l, ks_h, ks_l, vt_h, vt_l, qmask,
      preW, preB, postW, postB, rel, asub, ao_h, ao_l);

  proj_gemm<2><<<ggrid, gblk, 0, stream>>>(ao_h, ao_l, wo_h, wo_l,
                                           (float*)d_out, nullptr, nullptr);
}
